// LoRALinear_67130338836603
// MI455X (gfx1250) — matrix operand order repaired and checked by reading, not compiled, not run on hardware
//
#include <hip/hip_runtime.h>

typedef __attribute__((ext_vector_type(16))) __bf16 v16bf;
typedef __attribute__((ext_vector_type(8)))  float  v8f;
typedef unsigned short ushort_t;

#define M_DIM 16384
#define N_DIM 4096
#define K_DIM 4096
#define R_DIM 16
#define LORA_SCALE 1.0f   // alpha / r = 16 / 16

#define BLK 128            // M and N tile per workgroup
#define KT  32             // K per stage (one bf16 WMMA k-step)
#define KP  40             // padded LDS row stride (ushorts): 80B
#define NKMAIN (K_DIM / KT)    // 128 main K tiles
#define NITER  (NKMAIN + 1)    // + 1 LoRA (rank-16, padded) tile
#define ARR_USH   (BLK * KP)
#define STAGE_USH (4 * ARR_USH)
#define LDS_BYTES_FAST (3 * STAGE_USH * 2)   // 122880 B, triple buffered (async)
#define LDS_BYTES_FB   (2 * STAGE_USH * 2)   // 81920 B, double buffered (fallback)

struct alignas(16) F4 { float x, y, z, w; };
struct alignas(16) U4 { unsigned a, b, c, d; };
struct alignas(8)  US4 { ushort_t a, b, c, d; };
union Frag { U4 q[2]; v16bf v; };

// ---- CDNA5 async global->LDS copy (ASYNCcnt), zero data VGPRs ----
__device__ __forceinline__ void async_load_b128(unsigned lds_addr,
                                                const void* sbase,
                                                unsigned voff_bytes) {
  asm volatile("global_load_async_to_lds_b128 %0, %1, %2"
               :: "v"(lds_addr), "v"(voff_bytes),
                  "s"((unsigned long long)(__SIZE_TYPE__)sbase)
               : "memory");
}
__device__ __forceinline__ void wait_async_le8() {   // oldest stage done
  asm volatile("s_wait_asynccnt 0x8" ::: "memory");
}
__device__ __forceinline__ void wait_async_le0() {   // all stages done
  asm volatile("s_wait_asynccnt 0x0" ::: "memory");
}

// Truncation-based fp32 -> (bf16 hi, bf16 lo) split.
// hi = trunc16(x); r = x - hi is EXACT (Sterbenz); lo = trunc16(r).
__device__ __forceinline__ void cvt_split2(float f, ushort_t& h, ushort_t& l) {
  unsigned u = __float_as_uint(f);
  h = (ushort_t)(u >> 16);
  float r = f - __uint_as_float(u & 0xffff0000u);
  l = (ushort_t)(__float_as_uint(r) >> 16);
}

// ---------------- Kernel 1: xa = SCALE * (x @ A^T), [M,16] fp32 ----------------
__global__ __launch_bounds__(256) void lora_xa_kernel(
    const float* __restrict__ X, const float* __restrict__ LA,
    float* __restrict__ XA) {
  const int lane = threadIdx.x & 31;
  const int wid  = threadIdx.x >> 5;
  const int m    = blockIdx.x * 8 + wid;    // one wave per row
  float acc[16];
#pragma unroll
  for (int r = 0; r < 16; ++r) acc[r] = 0.f;
  const float* xr = X + (size_t)m * K_DIM;
  for (int k0 = lane * 4; k0 < K_DIM; k0 += 128) {
    F4 xv = *(const F4*)(xr + k0);
#pragma unroll
    for (int r = 0; r < 16; ++r) {
      F4 av = *(const F4*)(LA + r * K_DIM + k0);
      acc[r] += xv.x * av.x + xv.y * av.y + xv.z * av.z + xv.w * av.w;
    }
  }
#pragma unroll
  for (int r = 0; r < 16; ++r) {
#pragma unroll
    for (int off = 16; off >= 1; off >>= 1)
      acc[r] += __shfl_xor(acc[r], off, 32);
  }
  if (lane == 0) {
#pragma unroll
    for (int r = 0; r < 16; ++r)
      XA[(size_t)m * R_DIM + r] = LORA_SCALE * acc[r];
  }
}

// ---------- Pre-convert: fp32 array -> bf16 hi/lo planes (memory bound) ----------
__global__ __launch_bounds__(256) void cvt_planes_kernel(
    const float* __restrict__ SRC, ushort_t* __restrict__ HI,
    ushort_t* __restrict__ LO, long n4) {
  long stride = (long)gridDim.x * blockDim.x;
  for (long i = (long)blockIdx.x * blockDim.x + threadIdx.x; i < n4; i += stride) {
    F4 v = ((const F4*)SRC)[i];
    US4 h, l;
    cvt_split2(v.x, h.a, l.a); cvt_split2(v.y, h.b, l.b);
    cvt_split2(v.z, h.c, l.c); cvt_split2(v.w, h.d, l.d);
    ((US4*)HI)[i] = h;
    ((US4*)LO)[i] = l;
  }
}

// ---- Pre-convert + pad rank-16 fp32 [rows,16] -> bf16 hi/lo [rows,32] (zeros) ----
__global__ __launch_bounds__(256) void pad16_planes_kernel(
    const float* __restrict__ SRC, ushort_t* __restrict__ HI,
    ushort_t* __restrict__ LO, int rows) {
  int i = blockIdx.x * blockDim.x + threadIdx.x;    // one float4 (=4 k) per thread
  if (i >= rows * 8) return;
  int row = i >> 3, q = i & 7;
  F4 v = (q < 4) ? *(const F4*)(SRC + (size_t)row * R_DIM + q * 4)
                 : F4{0.f, 0.f, 0.f, 0.f};
  US4 h, l;
  cvt_split2(v.x, h.a, l.a); cvt_split2(v.y, h.b, l.b);
  cvt_split2(v.z, h.c, l.c); cvt_split2(v.w, h.d, l.d);
  *(US4*)(HI + (size_t)row * KT + q * 4) = h;
  *(US4*)(LO + (size_t)row * KT + q * 4) = l;
}

// --------- shared WMMA stage compute (ISA 7.12.2 wave32 fragment layouts) ---------
__device__ __forceinline__ void wmma_compute_stage(
    const ushort_t* st, int lane, int wid, v8f (&acc)[2][4]) {
  const ushort_t* Ah = st;
  const ushort_t* Al = st + ARR_USH;
  const ushort_t* Bh = st + 2 * ARR_USH;
  const ushort_t* Bl = st + 3 * ARR_USH;
  const int akb  = (lane >> 4) * 8;                 // A: K {b..b+7, b+16..b+23}
  const int bkb  = (lane >> 4) * 8;                 // B: K {b..b+7, b+16..b+23}
  const int mrow = (wid >> 1) * 32 + (lane & 15);
  const int nrow = (wid & 1) * 64 + (lane & 15);

  Frag bh[4], bl[4];
#pragma unroll
  for (int tn = 0; tn < 4; ++tn) {
    int r = nrow + tn * 16;
    bh[tn].q[0] = *(const U4*)(Bh + r * KP + bkb);
    bh[tn].q[1] = *(const U4*)(Bh + r * KP + bkb + 16);
    bl[tn].q[0] = *(const U4*)(Bl + r * KP + bkb);
    bl[tn].q[1] = *(const U4*)(Bl + r * KP + bkb + 16);
  }
#pragma unroll
  for (int tm = 0; tm < 2; ++tm) {
    int r = mrow + tm * 16;
    Frag ah, al;
    ah.q[0] = *(const U4*)(Ah + r * KP + akb);
    ah.q[1] = *(const U4*)(Ah + r * KP + akb + 16);
    al.q[0] = *(const U4*)(Al + r * KP + akb);
    al.q[1] = *(const U4*)(Al + r * KP + akb + 16);
#pragma unroll
    for (int tn = 0; tn < 4; ++tn) {     // bf16x3: hi*hi + hi*lo + lo*hi
      acc[tm][tn] = __builtin_amdgcn_wmma_f32_16x16x32_bf16(
          false, ah.v, false, bh[tn].v, (short)0, acc[tm][tn], false, false);
      acc[tm][tn] = __builtin_amdgcn_wmma_f32_16x16x32_bf16(
          false, ah.v, false, bl[tn].v, (short)0, acc[tm][tn], false, false);
      acc[tm][tn] = __builtin_amdgcn_wmma_f32_16x16x32_bf16(
          false, al.v, false, bh[tn].v, (short)0, acc[tm][tn], false, false);
    }
  }
}

__device__ __forceinline__ void wmma_epilogue(
    const float* __restrict__ BIAS, float* __restrict__ OUT,
    int Mblk, int Nblk, int lane, int wid, v8f (&acc)[2][4]) {
#pragma unroll
  for (int tn = 0; tn < 4; ++tn) {
    int n = Nblk + (wid & 1) * 64 + tn * 16 + (lane & 15);
    float bv = BIAS[n];
#pragma unroll
    for (int tm = 0; tm < 2; ++tm) {
      int mbase = Mblk + (wid >> 1) * 32 + tm * 16 + (lane >> 4) * 8;
#pragma unroll
      for (int i = 0; i < 8; ++i)
        OUT[(size_t)(mbase + i) * N_DIM + n] = acc[tm][tn][i] + bv;
    }
  }
}

// ======== FAST GEMM: bf16 hi/lo planes + async global->LDS, triple buffered ========
__global__ __launch_bounds__(256) void lora_gemm_pre_kernel(
    const ushort_t* __restrict__ XH, const ushort_t* __restrict__ XL,
    const ushort_t* __restrict__ WH, const ushort_t* __restrict__ WL,
    const ushort_t* __restrict__ XAH, const ushort_t* __restrict__ XAL,
    const ushort_t* __restrict__ LBH, const ushort_t* __restrict__ LBL,
    const float* __restrict__ BIAS, float* __restrict__ OUT) {
  extern __shared__ ushort_t smem[];
  const int tid  = threadIdx.x;
  const int lane = tid & 31;
  const int wid  = tid >> 5;
  const int Mblk = blockIdx.y * BLK;
  const int Nblk = blockIdx.x * BLK;
  const unsigned lds_base = (unsigned)(__SIZE_TYPE__)smem;  // LDS byte offset

  v8f acc[2][4];
#pragma unroll
  for (int tm = 0; tm < 2; ++tm)
#pragma unroll
    for (int tn = 0; tn < 4; ++tn) acc[tm][tn] = v8f{};

  // Each thread issues exactly 8 async b128 loads per stage (uniform ASYNCcnt).
  auto issue_async = [&](int it) {
    const ushort_t *pah, *pal, *pbh, *pbl;
    unsigned stA, stB;  // byte strides between rows
    if (it < NKMAIN) {
      size_t off = (size_t)it * KT;
      pah = XH + (size_t)Mblk * K_DIM + off;
      pal = XL + (size_t)Mblk * K_DIM + off;
      pbh = WH + (size_t)Nblk * K_DIM + off;
      pbl = WL + (size_t)Nblk * K_DIM + off;
      stA = K_DIM * 2; stB = K_DIM * 2;
    } else {            // rank-16 LoRA tile, already zero-padded to K=32
      pah = XAH + (size_t)Mblk * KT;  pal = XAL + (size_t)Mblk * KT;
      pbh = LBH + (size_t)Nblk * KT;  pbl = LBL + (size_t)Nblk * KT;
      stA = KT * 2; stB = KT * 2;
    }
    unsigned sb = lds_base + (unsigned)(it % 3) * (STAGE_USH * 2);
#pragma unroll
    for (int j = 0; j < 2; ++j) {
      int cu = tid + j * 256;            // 512 16B chunks per array (128 rows x 4)
      int row = cu >> 2, c = cu & 3;
      unsigned lOff = (unsigned)row * (KP * 2) + c * 16;
      unsigned gOffA = (unsigned)row * stA + c * 16;
      unsigned gOffB = (unsigned)row * stB + c * 16;
      async_load_b128(sb + 0 * (ARR_USH * 2) + lOff, pah, gOffA);
      async_load_b128(sb + 1 * (ARR_USH * 2) + lOff, pal, gOffA);
      async_load_b128(sb + 2 * (ARR_USH * 2) + lOff, pbh, gOffB);
      async_load_b128(sb + 3 * (ARR_USH * 2) + lOff, pbl, gOffB);
    }
  };

  issue_async(0);
  issue_async(1);                       // 2 stages in flight
  for (int it = 0; it < NITER; ++it) {
    if (it + 1 < NITER) wait_async_le8();  // oldest (stage it) landed
    else                wait_async_le0();
    __syncthreads();                       // publish all waves' stage-it writes
    wmma_compute_stage(smem + (it % 3) * STAGE_USH, lane, wid, acc);
    if (it + 2 < NITER) issue_async(it + 2);  // refill pipeline behind compute
  }
  wmma_epilogue(BIAS, OUT, Mblk, Nblk, lane, wid, acc);
}

// ======== FALLBACK GEMM: fp32 loads + in-loop truncation split (small d_ws) ========
__global__ __launch_bounds__(256) void lora_gemm_kernel(
    const float* __restrict__ X, const float* __restrict__ W,
    const float* __restrict__ BIAS, const float* __restrict__ XA,
    const float* __restrict__ LB, float* __restrict__ OUT) {
  extern __shared__ ushort_t smem[];
  const int tid  = threadIdx.x;
  const int lane = tid & 31;
  const int wid  = tid >> 5;
  const int Mblk = blockIdx.y * BLK;
  const int Nblk = blockIdx.x * BLK;

  v8f acc[2][4];
#pragma unroll
  for (int tm = 0; tm < 2; ++tm)
#pragma unroll
    for (int tn = 0; tn < 4; ++tn) acc[tm][tn] = v8f{};

  F4 ra[4], rb[4];

  auto loadg = [&](int it) {
    if (it < NKMAIN) {
      const float* gx = X + (size_t)Mblk * K_DIM + (size_t)it * KT;
      const float* gw = W + (size_t)Nblk * K_DIM + (size_t)it * KT;
#pragma unroll
      for (int j = 0; j < 4; ++j) {
        int idx = tid + j * 256;
        int row = idx >> 3, c4 = idx & 7;
        ra[j] = *(const F4*)(gx + (size_t)row * K_DIM + c4 * 4);
        rb[j] = *(const F4*)(gw + (size_t)row * K_DIM + c4 * 4);
      }
    } else {
#pragma unroll
      for (int j = 0; j < 4; ++j) {
        int idx = tid + j * 256;
        int row = idx >> 3, c4 = idx & 7;
        if (c4 < 4) {
          ra[j] = *(const F4*)(XA + (size_t)(Mblk + row) * R_DIM + c4 * 4);
          rb[j] = *(const F4*)(LB + (size_t)(Nblk + row) * R_DIM + c4 * 4);
        } else {
          ra[j] = F4{0.f, 0.f, 0.f, 0.f};
          rb[j] = F4{0.f, 0.f, 0.f, 0.f};
        }
      }
    }
  };

  auto stage_lds = [&](ushort_t* st) {
    ushort_t* Ah = st;
    ushort_t* Al = st + ARR_USH;
    ushort_t* Bh = st + 2 * ARR_USH;
    ushort_t* Bl = st + 3 * ARR_USH;
#pragma unroll
    for (int j = 0; j < 4; ++j) {
      int idx = tid + j * 256;
      int row = idx >> 3, c4 = idx & 7;
      US4 h, l;
      cvt_split2(ra[j].x, h.a, l.a); cvt_split2(ra[j].y, h.b, l.b);
      cvt_split2(ra[j].z, h.c, l.c); cvt_split2(ra[j].w, h.d, l.d);
      *(US4*)(Ah + row * KP + c4 * 4) = h;
      *(US4*)(Al + row * KP + c4 * 4) = l;
      cvt_split2(rb[j].x, h.a, l.a); cvt_split2(rb[j].y, h.b, l.b);
      cvt_split2(rb[j].z, h.c, l.c); cvt_split2(rb[j].w, h.d, l.d);
      *(US4*)(Bh + row * KP + c4 * 4) = h;
      *(US4*)(Bl + row * KP + c4 * 4) = l;
    }
  };

  loadg(0);
  stage_lds(smem);
  __syncthreads();
  for (int it = 0; it < NITER; ++it) {
    if (it + 1 < NITER) loadg(it + 1);
    wmma_compute_stage(smem + (it & 1) * STAGE_USH, lane, wid, acc);
    if (it + 1 < NITER) stage_lds(smem + ((it + 1) & 1) * STAGE_USH);
    __syncthreads();
  }
  wmma_epilogue(BIAS, OUT, Mblk, Nblk, lane, wid, acc);
}

extern "C" void kernel_launch(void* const* d_in, const int* in_sizes, int n_in,
                              void* d_out, int out_size, void* d_ws, size_t ws_size,
                              hipStream_t stream) {
  const float* x    = (const float*)d_in[0];   // [4,4096,4096]
  const float* w    = (const float*)d_in[1];   // [4096,4096]
  const float* bias = (const float*)d_in[2];   // [4096]
  const float* lA   = (const float*)d_in[3];   // [16,4096]
  const float* lB   = (const float*)d_in[4];   // [4096,16]
  float* out = (float*)d_out;

  const size_t SZ_XAF = (size_t)M_DIM * R_DIM * 4;       // fp32 xa
  const size_t SZ_XP  = (size_t)M_DIM * K_DIM * 2;       // x hi or lo plane
  const size_t SZ_WP  = (size_t)N_DIM * K_DIM * 2;       // W hi or lo plane
  const size_t SZ_XAP = (size_t)M_DIM * KT * 2;          // padded xa plane
  const size_t SZ_LBP = (size_t)N_DIM * KT * 2;          // padded lora_B plane
  char* ws = (char*)d_ws;
  float*    xaf = (float*)ws;
  ushort_t* xh  = (ushort_t*)(ws + SZ_XAF);
  ushort_t* xl  = (ushort_t*)(ws + SZ_XAF + SZ_XP);
  ushort_t* wh  = (ushort_t*)(ws + SZ_XAF + 2 * SZ_XP);
  ushort_t* wl  = (ushort_t*)(ws + SZ_XAF + 2 * SZ_XP + SZ_WP);
  ushort_t* xah = (ushort_t*)(ws + SZ_XAF + 2 * SZ_XP + 2 * SZ_WP);
  ushort_t* xal = (ushort_t*)(ws + SZ_XAF + 2 * SZ_XP + 2 * SZ_WP + SZ_XAP);
  ushort_t* lbh = (ushort_t*)(ws + SZ_XAF + 2 * SZ_XP + 2 * SZ_WP + 2 * SZ_XAP);
  ushort_t* lbl = (ushort_t*)(ws + SZ_XAF + 2 * SZ_XP + 2 * SZ_WP + 2 * SZ_XAP + SZ_LBP);
  const size_t NEED_FAST =
      SZ_XAF + 2 * SZ_XP + 2 * SZ_WP + 2 * SZ_XAP + 2 * SZ_LBP;  // ~324 MB

  dim3 grid(N_DIM / BLK, M_DIM / BLK);                   // (32, 128)
  lora_xa_kernel<<<M_DIM / 8, 256, 0, stream>>>(x, lA, xaf);

  if (ws_size >= NEED_FAST) {
    long n4x = (long)M_DIM * K_DIM / 4, n4w = (long)N_DIM * K_DIM / 4;
    cvt_planes_kernel<<<4096, 256, 0, stream>>>(x, xh, xl, n4x);
    cvt_planes_kernel<<<2048, 256, 0, stream>>>(w, wh, wl, n4w);
    pad16_planes_kernel<<<(M_DIM * 8 + 255) / 256, 256, 0, stream>>>(xaf, xah, xal, M_DIM);
    pad16_planes_kernel<<<(N_DIM * 8 + 255) / 256, 256, 0, stream>>>(lB, lbh, lbl, N_DIM);
    lora_gemm_pre_kernel<<<grid, 256, LDS_BYTES_FAST, stream>>>(
        xh, xl, wh, wl, xah, xal, lbh, lbl, bias, out);
  } else {
    lora_gemm_kernel<<<grid, 256, LDS_BYTES_FB, stream>>>(x, w, bias, xaf, lB, out);
  }
}
